// ScaledDotProductAttention_43585328120083
// MI455X (gfx1250) — matrix operand order repaired and checked by reading, not compiled, not run on hardware
//
#include <hip/hip_runtime.h>

// =====================================================================
// AutoCorrelation (Autoformer-style) for MI455X / gfx1250, wave32.
//
//   corr[row,d] = sum_e q[row,e] * k[row,(e-d) mod 64]        (exact: DFT domain)
//   top-4(corr) -> softmax -> V[row,e] = sum_i w_i * v[row,(e+d_i) mod 64]
//   out = [ V (B,H,L,E) , corr^T (B,E,H,L) ]  flat f32
//
// The circular correlation is evaluated as GEMMs against *shared* 64x64
// cos/sin DFT matrices so it runs on the WMMA pipe:
//   Qre=q*C, Qs=q*S, Kre=k*C, Ks=k*S
//   Rre' = (Qre*Kre + Qs*Ks)/64 ;  Rim' = (Qs*Kre - Qre*Ks)/64
//   corr = Rre'*C + Rim'*S          (C,S symmetric -> same frag tables)
// Inputs are split into bf16 hi/lo and each logical GEMM uses the 3-term
// split (ah*bh + ah*bl + al*bh) for ~f32 accuracy on v_wmma_f32_16x16x32_bf16.
// The values tile is streamed global->LDS with the CDNA5 async path
// (GLOBAL_LOAD_ASYNC_TO_LDS_B128 + s_wait_asynccnt) when available, so it
// overlaps the whole WMMA pipeline.
// =====================================================================

typedef float  v8f   __attribute__((ext_vector_type(8)));
typedef __bf16 v16bf __attribute__((ext_vector_type(16)));
typedef __bf16 v8bf  __attribute__((ext_vector_type(8)));

#define NW  4            // waves per workgroup
#define TPW 8            // 16-row tiles per wave

#if defined(__AMDGCN__) && __has_builtin(__builtin_amdgcn_global_load_async_to_lds_b128) && __has_builtin(__builtin_amdgcn_s_wait_asynccnt)
#define USE_ASYNC 1
#define AS1 __attribute__((address_space(1)))
#define AS3 __attribute__((address_space(3)))
typedef int v4i __attribute__((vector_size(4 * sizeof(int))));  // matches builtin param
#else
#define USE_ASYNC 0
#endif

constexpr int Bn = 32, Hn = 8, Ln = 2048, En = 64;
constexpr int NROWS  = Bn * Hn * Ln;        // 524288
constexpr int NTILES = NROWS / 16;          // 32768
constexpr int VOUT_ELEMS = Bn * Hn * Ln * En;

__device__ __forceinline__ unsigned short f2bf(float x) {
  unsigned u = __float_as_uint(x);
  unsigned r = u + 0x7FFFu + ((u >> 16) & 1u);   // round-to-nearest-even
  return (unsigned short)(r >> 16);
}
__device__ __forceinline__ float bf2f(unsigned short h) {
  return __uint_as_float(((unsigned)h) << 16);
}

// A-fragment (16x32 bf16, documented CDNA5 layout): per lane two 8-element
// contiguous K runs, 16 K apart.  p must point at row base + 8*(lane>=16).
__device__ __forceinline__ v16bf load_a(const __bf16* p, int ks) {
  v8bf lo = *(const v8bf*)(p + ks * 32);
  v8bf hi = *(const v8bf*)(p + ks * 32 + 16);
  return __builtin_shufflevector(lo, hi, 0,1,2,3,4,5,6,7,8,9,10,11,12,13,14,15);
}

__device__ __forceinline__ v8f wmma_bf(v16bf a, v16bf b, v8f c) {
  return __builtin_amdgcn_wmma_f32_16x16x32_bf16(false, a, false, b, (short)0, c,
                                                 false, false);
}

__device__ __forceinline__ void split_store4(unsigned short* hd, unsigned short* ld, float4 x) {
  unsigned short h0 = f2bf(x.x), h1 = f2bf(x.y), h2 = f2bf(x.z), h3 = f2bf(x.w);
  unsigned short l0 = f2bf(x.x - bf2f(h0)), l1 = f2bf(x.y - bf2f(h1));
  unsigned short l2 = f2bf(x.z - bf2f(h2)), l3 = f2bf(x.w - bf2f(h3));
  uint2 ph, pl;
  ph.x = (unsigned)h0 | ((unsigned)h1 << 16); ph.y = (unsigned)h2 | ((unsigned)h3 << 16);
  pl.x = (unsigned)l0 | ((unsigned)l1 << 16); pl.y = (unsigned)l2 | ((unsigned)l3 << 16);
  *(uint2*)hd = ph;
  *(uint2*)ld = pl;
}

__global__ __launch_bounds__(NW * 32) void
autocorr_kernel(const float* __restrict__ Qg, const float* __restrict__ Kg,
                const float* __restrict__ Vg, float* __restrict__ outV,
                float* __restrict__ outC) {
  // ---- LDS ----------------------------------------------------------
  // DFT B-fragment tables (hi/lo bf16), pre-swizzled into the 16-bit
  // 32x16 B layout: frag = ks*4 + nt; lane<16 -> K 0..7,16..23, lane>=16 -> K 8..15,24..31,
  // element i -> VGPR i/2 half i%2; N = 16*nt + lane%16; K_global = 32*ks + ...
  __shared__ __align__(32) unsigned short tabCh[8][32][16];
  __shared__ __align__(32) unsigned short tabCl[8][32][16];
  __shared__ __align__(32) unsigned short tabSh[8][32][16];
  __shared__ __align__(32) unsigned short tabSl[8][32][16];
  __shared__ float basec[64];
  // per-wave: [mat][row][72] bf16 rows (stride 72 = conflict-tuned, 144B)
  __shared__ __align__(16) unsigned short grpA[NW][4][16][72];  // qh,ql,kh,kl / later corr,wd
  __shared__ __align__(16) unsigned short grpB[NW][4][16][72];  // rreh,rrel,rimh,riml / merge buf
  __shared__ __align__(16) float vstg[NW][16 * 68];             // values tile (async target)

  const int tid  = threadIdx.x;
  const int lane = tid & 31;
  const int wid  = tid >> 5;
  const int r    = lane & 15;   // row-in-tile (M) / n index
  const int hl   = lane >> 4;   // high-lane flag

  // ---- build DFT tables (once per block) ----------------------------
  const float W = 6.28318530717958647692f / 64.0f;
  for (int k2 = tid; k2 < 64; k2 += NW * 32) basec[k2] = cosf(W * (float)k2);
  __syncthreads();
  for (int s = tid; s < 256; s += NW * 32) {   // (frag, lane) slots
    int fr = s >> 5, ln = s & 31;
    int Nc = 16 * (fr & 3) + (ln & 15);
    int Kb = 32 * (fr >> 2) + 8 * (ln >> 4);
#pragma unroll
    for (int i = 0; i < 16; ++i) {
      int kn   = ((Kb + i + (i & 8)) * Nc) & 63;
      float c  = basec[kn];
      float sv = basec[(kn + 48) & 63];        // sin(x) = cos(x - pi/2)
      unsigned short ch = f2bf(c);
      tabCh[fr][ln][i] = ch;  tabCl[fr][ln][i] = f2bf(c - bf2f(ch));
      unsigned short sh = f2bf(sv);
      tabSh[fr][ln][i] = sh;  tabSl[fr][ln][i] = f2bf(sv - bf2f(sh));
    }
  }
  __syncthreads();

  // phase-3 aliases over grpA (q/k splits are dead by then)
  float* v_lds    = &vstg[wid][0];                                 // 16 x 68 f32
  float* corr_lds = (float*)((char*)&grpA[wid][0][0][0] + 4352);   // 16 x 68 f32
  float* wdw      = (float*)((char*)&grpA[wid][0][0][0] + 8704);   // 16 x (4 w + 4 idx)
  int*   wdi      = (int*)  ((char*)&grpA[wid][0][0][0] + 8704);
  float* mrgV     = (float*)&grpB[wid][0][0][0];                   // 16 x 2 x 4
  int*   mrgI     = (int*)((char*)&grpB[wid][0][0][0] + 512);

  const int tile0 = (blockIdx.x * NW + wid) * TPW;
  const v8f Zero = {0, 0, 0, 0, 0, 0, 0, 0};

#pragma unroll 1
  for (int tt = 0; tt < TPW; ++tt) {
    const int tile = tile0 + tt;
    const float* qg = Qg + (size_t)tile * 1024;
    const float* kg = Kg + (size_t)tile * 1024;
    const float* vg = Vg + (size_t)tile * 1024;
    __builtin_prefetch(qg + 1024, 0, 1);
    __builtin_prefetch(kg + 1024, 0, 1);

#if USE_ASYNC
    // ---- kick off async global->LDS copy of the values tile; it
    // overlaps the whole forward/inverse WMMA pipeline below ----------
#pragma unroll
    for (int j = 0; j < 8; ++j) {
      int idx = lane * 4 + j * 128;
      int row = idx >> 6, e = idx & 63;
      __builtin_amdgcn_global_load_async_to_lds_b128(
          (AS1 v4i*)(vg + idx), (AS3 v4i*)&v_lds[row * 68 + e], 0, 0);
    }
#endif

    // ---- load q,k tile; split to bf16 hi/lo rows in LDS -------------
#pragma unroll
    for (int j = 0; j < 8; ++j) {
      int idx = lane * 4 + j * 128;
      int row = idx >> 6, e = idx & 63;
      float4 xq = *(const float4*)(qg + idx);
      float4 xk = *(const float4*)(kg + idx);
      split_store4(&grpA[wid][0][row][e], &grpA[wid][1][row][e], xq);
      split_store4(&grpA[wid][2][row][e], &grpA[wid][3][row][e], xk);
    }

    const __bf16* qhp = (const __bf16*)&grpA[wid][0][r][8 * hl];
    const __bf16* qlp = (const __bf16*)&grpA[wid][1][r][8 * hl];
    const __bf16* khp = (const __bf16*)&grpA[wid][2][r][8 * hl];
    const __bf16* klp = (const __bf16*)&grpA[wid][3][r][8 * hl];

    // ---- forward DFT per 16-frequency tile; store R to bounce -------
#pragma unroll
    for (int nt = 0; nt < 4; ++nt) {
      v8f Qre = Zero, Qs = Zero, Kre = Zero, Ks = Zero;
#pragma unroll
      for (int ks = 0; ks < 2; ++ks) {
        v16bf Aqh = load_a(qhp, ks), Aql = load_a(qlp, ks);
        v16bf Akh = load_a(khp, ks), Akl = load_a(klp, ks);
        int fr = ks * 4 + nt;
        v16bf Bch = *(const v16bf*)&tabCh[fr][lane][0];
        v16bf Bcl = *(const v16bf*)&tabCl[fr][lane][0];
        v16bf Bsh = *(const v16bf*)&tabSh[fr][lane][0];
        v16bf Bsl = *(const v16bf*)&tabSl[fr][lane][0];
        Qre = wmma_bf(Aqh, Bch, Qre); Qre = wmma_bf(Aqh, Bcl, Qre); Qre = wmma_bf(Aql, Bch, Qre);
        Qs  = wmma_bf(Aqh, Bsh, Qs ); Qs  = wmma_bf(Aqh, Bsl, Qs ); Qs  = wmma_bf(Aql, Bsh, Qs );
        Kre = wmma_bf(Akh, Bch, Kre); Kre = wmma_bf(Akh, Bcl, Kre); Kre = wmma_bf(Akl, Bch, Kre);
        Ks  = wmma_bf(Akh, Bsh, Ks ); Ks  = wmma_bf(Akh, Bsl, Ks ); Ks  = wmma_bf(Akl, Bsh, Ks );
      }
      // res = QF * conj(KF); fold 1/64 and the inverse's sign here.
      v8f Rre = (Qre * Kre + Qs * Ks) * 0.015625f;
      v8f Rim = (Qs * Kre - Qre * Ks) * 0.015625f;
#pragma unroll
      for (int i = 0; i < 8; ++i) {          // C/D layout: m = i + 8*hl, n = r
        int m = i + 8 * hl, f = nt * 16 + r;
        float re = Rre[i];
        unsigned short rh = f2bf(re);
        grpB[wid][0][m][f] = rh;  grpB[wid][1][m][f] = f2bf(re - bf2f(rh));
        float im = Rim[i];
        unsigned short ih = f2bf(im);
        grpB[wid][2][m][f] = ih;  grpB[wid][3][m][f] = f2bf(im - bf2f(ih));
      }
    }

    // ---- inverse DFT: corr = Rre'*C + Rim'*S ------------------------
    v8f corr[4];
#pragma unroll
    for (int dt = 0; dt < 4; ++dt) corr[dt] = Zero;
    const __bf16* rhp = (const __bf16*)&grpB[wid][0][r][8 * hl];
    const __bf16* rlp = (const __bf16*)&grpB[wid][1][r][8 * hl];
    const __bf16* ihp = (const __bf16*)&grpB[wid][2][r][8 * hl];
    const __bf16* ilp = (const __bf16*)&grpB[wid][3][r][8 * hl];
#pragma unroll
    for (int ks = 0; ks < 2; ++ks) {
      v16bf Arh = load_a(rhp, ks), Arl = load_a(rlp, ks);
      v16bf Aih = load_a(ihp, ks), Ail = load_a(ilp, ks);
#pragma unroll
      for (int dt = 0; dt < 4; ++dt) {
        int fr = ks * 4 + dt;                 // C,S symmetric -> same tables
        v16bf Bch = *(const v16bf*)&tabCh[fr][lane][0];
        v16bf Bcl = *(const v16bf*)&tabCl[fr][lane][0];
        v16bf Bsh = *(const v16bf*)&tabSh[fr][lane][0];
        v16bf Bsl = *(const v16bf*)&tabSl[fr][lane][0];
        corr[dt] = wmma_bf(Arh, Bch, corr[dt]); corr[dt] = wmma_bf(Arh, Bcl, corr[dt]);
        corr[dt] = wmma_bf(Arl, Bch, corr[dt]);
        corr[dt] = wmma_bf(Aih, Bsh, corr[dt]); corr[dt] = wmma_bf(Aih, Bsl, corr[dt]);
        corr[dt] = wmma_bf(Ail, Bsh, corr[dt]);
      }
    }

#if !USE_ASYNC
    // ---- synchronous fallback: load v tile into LDS -----------------
#pragma unroll
    for (int j = 0; j < 8; ++j) {
      int idx = lane * 4 + j * 128;
      int row = idx >> 6, e = idx & 63;
      float4 xv = *(const float4*)(vg + idx);
      *(float4*)&v_lds[row * 68 + e] = xv;
    }
#endif

    // ---- stage corr to LDS ------------------------------------------
#pragma unroll
    for (int dt = 0; dt < 4; ++dt)
#pragma unroll
      for (int i = 0; i < 8; ++i)
        corr_lds[(i + 8 * hl) * 68 + dt * 16 + r] = corr[dt][i];

    // ---- top-4 per row (2 lanes/row, branchless insert), merge, softmax
    {
      float t0 = -3.4e38f, t1 = t0, t2 = t0, t3 = t0;
      int   j0 = 0, j1 = 0, j2 = 0, j3 = 0;
      int dbase = hl * 32;
      for (int s = 0; s < 32; ++s) {
        int d = dbase + s;
        float c = corr_lds[r * 68 + d];
        bool b0 = c > t0, b1 = c > t1, b2 = c > t2, b3 = c > t3;
        t3 = b2 ? t2 : (b3 ? c : t3);  j3 = b2 ? j2 : (b3 ? d : j3);
        t2 = b1 ? t1 : (b2 ? c : t2);  j2 = b1 ? j1 : (b2 ? d : j2);
        t1 = b0 ? t0 : (b1 ? c : t1);  j1 = b0 ? j0 : (b1 ? d : j1);
        t0 = b0 ? c : t0;              j0 = b0 ? d : j0;
      }
      int mb = r * 8 + hl * 4;
      mrgV[mb + 0] = t0; mrgV[mb + 1] = t1; mrgV[mb + 2] = t2; mrgV[mb + 3] = t3;
      mrgI[mb + 0] = j0; mrgI[mb + 1] = j1; mrgI[mb + 2] = j2; mrgI[mb + 3] = j3;
      if (hl == 0) {
        float wv[4]; int wi[4];
        int pa = 0, pb = 0;
#pragma unroll
        for (int j = 0; j < 4; ++j) {
          float va = mrgV[r * 8 + pa], vb = mrgV[r * 8 + 4 + pb];
          bool ta = va >= vb;                   // ties -> lower delay (first half)
          wv[j] = ta ? va : vb;
          wi[j] = ta ? mrgI[r * 8 + pa] : mrgI[r * 8 + 4 + pb];
          pa += ta ? 1 : 0;  pb += ta ? 0 : 1;
        }
        float mx = wv[0];
        float e0 = __expf(wv[0] - mx), e1 = __expf(wv[1] - mx);
        float e2 = __expf(wv[2] - mx), e3 = __expf(wv[3] - mx);
        float inv = 1.0f / (e0 + e1 + e2 + e3);
        wdw[r * 8 + 0] = e0 * inv; wdw[r * 8 + 1] = e1 * inv;
        wdw[r * 8 + 2] = e2 * inv; wdw[r * 8 + 3] = e3 * inv;
        wdi[r * 8 + 4 + 0] = wi[0]; wdi[r * 8 + 4 + 1] = wi[1];
        wdi[r * 8 + 4 + 2] = wi[2]; wdi[r * 8 + 4 + 3] = wi[3];
      }
    }

    // ---- corr^T store: out[b,e,h,l], 16 consecutive l per chunk -----
    {
      int row0 = tile << 4;
      int bb = row0 >> 14;
      int hh = (row0 >> 11) & 7;
      int l0 = row0 & 2047;
      int cbase = ((bb * 64) * 8 + hh) * 2048 + l0;   // + e*16384 + l
#pragma unroll
      for (int t2 = 0; t2 < 32; ++t2) {
        int e = 2 * t2 + hl;
        outC[cbase + e * 16384 + r] = corr_lds[r * 68 + e];
      }
    }

#if USE_ASYNC
    __builtin_amdgcn_s_wait_asynccnt(0);   // v tile landed in LDS
#endif

    // ---- V = sum_i w_i * v[(e + d_i) mod 64] ------------------------
    {
      float* outVt = outV + (size_t)tile * 1024;
      int e0 = lane * 2;
#pragma unroll 1
      for (int rr = 0; rr < 16; ++rr) {
        const float* vrow = &v_lds[rr * 68];
        float a0 = 0.f, a1 = 0.f;
#pragma unroll
        for (int j = 0; j < 4; ++j) {
          float wj = wdw[rr * 8 + j];
          int   dj = wdi[rr * 8 + 4 + j];
          a0 += wj * vrow[(e0 + dj) & 63];
          a1 += wj * vrow[(e0 + 1 + dj) & 63];
        }
        float2 o; o.x = a0; o.y = a1;
        *(float2*)&outVt[rr * 64 + e0] = o;
      }
    }
  }
}

extern "C" void kernel_launch(void* const* d_in, const int* in_sizes, int n_in,
                              void* d_out, int out_size, void* d_ws, size_t ws_size,
                              hipStream_t stream) {
  const float* Q = (const float*)d_in[0];
  const float* K = (const float*)d_in[1];
  const float* V = (const float*)d_in[2];
  float* outV = (float*)d_out;
  float* outC = outV + (size_t)VOUT_ELEMS;

  dim3 grid(NTILES / (NW * TPW));   // 1024 blocks
  dim3 block(NW * 32);              // 4 waves (wave32)
  autocorr_kernel<<<grid, block, 0, stream>>>(Q, K, V, outV, outC);
}
